// ScalableViT_65180423685350
// MI455X (gfx1250) — matrix operand order repaired and checked by reading, not compiled, not run on hardware
//
#include <hip/hip_runtime.h>

typedef __attribute__((ext_vector_type(16))) __bf16       v16bf;
typedef __attribute__((ext_vector_type(8)))  float        v8f;
typedef __attribute__((ext_vector_type(4)))  unsigned int v4u;
typedef __attribute__((ext_vector_type(4)))  unsigned int u32x4;
typedef __attribute__((ext_vector_type(8)))  int          i32x8;
typedef __attribute__((ext_vector_type(4)))  int          i32x4;

#define Bb   64
#define Hh   56
#define Ww   56
#define Cc   128
#define NHh  4
#define HD   32
#define WSz  7
#define WIN  49
#define Nn   (Hh * Ww)           // 3136
#define Mtot (Bb * Nn)           // 200704

union Frag16 {
  v16bf v;
  v4u   q[2];
};

__device__ __forceinline__ unsigned short f2bf(float f) {
  unsigned int u = __float_as_uint(f);
  unsigned int r = (u + 0x7FFFu + ((u >> 16) & 1u)) >> 16;   // round-to-nearest-even
  return (unsigned short)r;
}
__device__ __forceinline__ float bf2f(unsigned short b) {
  return __uint_as_float(((unsigned int)b) << 16);
}

// A fragment (16x32 bf16, MxK). ISA 7.12.2: lane M = lane&15, half = lane>>4,
// VGPR 0-3 hold K = 2i,2i+1 (+8*half); VGPR 4-7 hold K = 16+2(i-4),+1 (+8*half).
__device__ __forceinline__ v16bf load_a_frag(const unsigned short* sm, int lda,
                                             int m0, int k0, int lane) {
  int m = m0 + (lane & 15);
  int half = (lane >> 4) & 1;
  const unsigned short* p = sm + m * lda + k0 + 8 * half;
  Frag16 f;
  f.q[0] = *(const v4u*)(p);
  f.q[1] = *(const v4u*)(p + 16);
  return f.v;
}

// B fragment (32x16 bf16, KxN) from LDS stored [n][k] (row stride ldb).
__device__ __forceinline__ v16bf load_b_frag(const unsigned short* sm, int ldb,
                                             int n0, int k0, int lane) {
  int n = n0 + (lane & 15);
  int half = (lane >> 4) & 1;
  const unsigned short* p = sm + n * ldb + k0 + 8 * half;
  Frag16 f;
  f.q[0] = *(const v4u*)(p);
  f.q[1] = *(const v4u*)(p + 16);
  return f.v;
}

__device__ __forceinline__ v8f wmma_bf16(v16bf a, v16bf b, v8f c) {
  return __builtin_amdgcn_wmma_f32_16x16x32_bf16(false, a, false, b, (short)0, c,
                                                 false, false);
}

// ---------------------------------------------------------------------------
// TDM descriptor builder: 3D tile (tile_d0 elems x tile_d1 x tile_d2), 2-byte
// elements, packed contiguously into LDS. Strides in element units.
// D# packing per CDNA5 ISA ch.8 (group0 128b, group1 256b, groups 2/3 128b).
// Toolchain here exposes the 6-arg builtin: (g0, g1, g2, g3, <x8 extra>, cpol);
// the extra int32x8 is passed zero-filled (probe-verified form).
// ---------------------------------------------------------------------------
__device__ __forceinline__ void tdm_load_tile_2B(unsigned lds_byte,
                                                 unsigned long long gaddr,
                                                 unsigned tile_d0, unsigned tile_d1,
                                                 unsigned tile_d2,
                                                 unsigned long long stride0,
                                                 unsigned long long stride1) {
  const unsigned td0 = 384u, td1 = 56u, td2 = 56u;  // tensor dims (never OOB here)
  u32x4 g0;
  g0[0] = 1u;                                        // count=1, user descriptor
  g0[1] = lds_byte;                                  // lds_addr [63:32]
  g0[2] = (unsigned)(gaddr & 0xFFFFFFFFull);         // global_addr lo
  g0[3] = (unsigned)((gaddr >> 32) & 0x1FFFFFFull)   // global_addr [56:32]
        | (2u << 30);                                // type=2 ("image")
  i32x8 g1;
  g1[0] = (int)(1u << 16);                           // wg_mask=0, data_size=1 (2B)
  g1[1] = (int)((td0 & 0xFFFFu) << 16);              // atomic_bar=0 | td0[15:0]
  g1[2] = (int)(((td0 >> 16) & 0xFFFFu) | ((td1 & 0xFFFFu) << 16));
  g1[3] = (int)(((td1 >> 16) & 0xFFFFu) | (tile_d0 << 16));
  g1[4] = (int)((tile_d1 & 0xFFFFu) | (tile_d2 << 16));
  g1[5] = (int)(stride0 & 0xFFFFFFFFull);            // dim0_stride [31:0]
  g1[6] = (int)(((stride0 >> 32) & 0xFFFFull) | ((stride1 & 0xFFFFull) << 16));
  g1[7] = (int)((stride1 >> 16) & 0xFFFFFFFFull);
  i32x4 g2;
  g2[0] = (int)td2;                                  // tensor_dim2
  g2[1] = 0;                                         // tensor_dim3
  g2[2] = 0;                                         // dim2_stride lo (unused)
  g2[3] = 0;                                         // dim2_stride hi | tile_dim3=0
  i32x4 g3;
  g3[0] = 0; g3[1] = 0; g3[2] = 0; g3[3] = 0;
  i32x8 gx;                                          // extra arg (zero-filled)
  gx[0] = 0; gx[1] = 0; gx[2] = 0; gx[3] = 0;
  gx[4] = 0; gx[5] = 0; gx[6] = 0; gx[7] = 0;
  __builtin_amdgcn_tensor_load_to_lds(g0, g1, g2, g3, gx, 0);
}

__device__ __forceinline__ unsigned lds_addr_of(const void* p) {
  // generic pointer to LDS: low 32 bits are the LDS byte offset (ISA 10.2)
  return (unsigned)(uintptr_t)p;
}

// ---------------------------------------------------------------------------
// Kernel 1: qkv = x @ w_qkv^T   (M=200704, K=128, N=384), bf16 out.
// Block: 128 threads (4 waves). Tile 64(M) x 64(N), full K staged once.
// Grid: (n-tiles, m-tiles) so the 6 blocks sharing an A tile are adjacent -> L2.
// ---------------------------------------------------------------------------
__global__ __launch_bounds__(128) void qkv_gemm(const float* __restrict__ x,
                                                const float* __restrict__ w,
                                                unsigned short* __restrict__ out) {
  __shared__ unsigned short smA[64 * 128];
  __shared__ unsigned short smW[64 * 128];
  const int tid = threadIdx.x;
  const int n0 = blockIdx.x * 64;
  const int m0 = blockIdx.y * 64;
  for (int i = tid; i < 64 * 128; i += 128) {
    int r = i >> 7, k = i & 127;
    smA[i] = f2bf(x[(size_t)(m0 + r) * 128 + k]);
    smW[i] = f2bf(w[(size_t)(n0 + r) * 128 + k]);
  }
  __syncthreads();
  const int wave = tid >> 5, lane = tid & 31;
  v8f acc[4] = {};
#pragma unroll
  for (int kt = 0; kt < 4; ++kt) {
    v16bf bf = load_b_frag(smW, 128, wave * 16, kt * 32, lane);
#pragma unroll
    for (int mi = 0; mi < 4; ++mi) {
      v16bf af = load_a_frag(smA, 128, mi * 16, kt * 32, lane);
      acc[mi] = wmma_bf16(af, bf, acc[mi]);
    }
  }
  const int jn = lane & 15, half = lane >> 4;
  const int col = n0 + wave * 16 + jn;
#pragma unroll
  for (int mi = 0; mi < 4; ++mi)
#pragma unroll
    for (int r = 0; r < 8; ++r) {
      int row = m0 + mi * 16 + r + 8 * half;
      out[(size_t)row * 384 + col] = f2bf(acc[mi][r]);
    }
}

// ---------------------------------------------------------------------------
// Kernel 2: depthwise 3x3 conv (SAME, zero pad) on V slice of qkv + bias.
// ---------------------------------------------------------------------------
__global__ __launch_bounds__(256) void lim_conv(const unsigned short* __restrict__ qkv,
                                                const float* __restrict__ wl,
                                                const float* __restrict__ bl,
                                                unsigned short* __restrict__ vlim) {
  long long i = (long long)blockIdx.x * 256 + threadIdx.x;
  int c = (int)(i & 127);
  int n = (int)((i >> 7) % Nn);
  int b = (int)(i / ((long long)Nn * 128));
  int h = n / Ww, w = n % Ww;
  float acc = bl[c];
#pragma unroll
  for (int dh = -1; dh <= 1; ++dh)
#pragma unroll
    for (int dw = -1; dw <= 1; ++dw) {
      int hh = h + dh, w2 = w + dw;
      if (hh >= 0 && hh < Hh && w2 >= 0 && w2 < Ww) {
        size_t idx = ((size_t)b * Nn + hh * Ww + w2) * 384 + 256 + c;
        acc += bf2f(qkv[idx]) * wl[c * 9 + (dh + 1) * 3 + (dw + 1)];
      }
    }
  vlim[i] = f2bf(acc);
}

// ---------------------------------------------------------------------------
// Kernel 3: windowed attention. Block = (window, b), 4 waves = 4 heads.
// Q/K tiles (49x32 bf16, strided 7x7 window) are DMA'd into LDS by the Tensor
// Data Mover (tensor_load_to_lds, TENSORcnt); V is staged transposed by VALU.
// S = Q K^T (16 WMMA), masked softmax via shuffles, P V (16 WMMA).
// ---------------------------------------------------------------------------
__global__ __launch_bounds__(128) void win_attn(const unsigned short* __restrict__ qkv,
                                                unsigned short* __restrict__ attnb) {
  __shared__ unsigned short sQ[NHh][64 * 32];
  __shared__ unsigned short sK[NHh][64 * 32];
  __shared__ unsigned short sVT[NHh][32 * 64];
  __shared__ unsigned short sP[NHh][16 * 64];

  const int head = threadIdx.x >> 5;
  const int lane = threadIdx.x & 31;
  const int g = blockIdx.x;                 // window 0..63
  const int b = blockIdx.y;
  const int gh = g >> 3, gw = g & 7;
  const int ho = head * HD;

  // Zero pad rows (tokens 49..63) of sQ/sK and stage V transposed.
  // These DS stores touch only regions disjoint from the TDM destinations.
  for (int idx = lane; idx < 64 * 32; idx += 32) {
    int t = idx >> 5, d = idx & 31;
    unsigned short vv = 0;
    if (t < WIN) {
      int h = gh * WSz + t / WSz;
      int w = gw * WSz + t % WSz;
      vv = qkv[((size_t)b * Nn + h * Ww + w) * 384 + 256 + ho + d];
    } else {
      sQ[head][idx] = 0;
      sK[head][idx] = 0;
    }
    sVT[head][d * 64 + t] = vv;
  }

  // TDM: Q and K window tiles -> LDS. Tile = 32 elems x 7 (W) x 7 (H),
  // element strides 384 (row) and 56*384 (image row). Lands as 49 contiguous
  // 64B rows in LDS = sQ/sK token-major layout. One issue per wave per matrix.
  {
    unsigned long long base =
        (unsigned long long)(uintptr_t)qkv +
        2ull * (((unsigned long long)b * Nn + (unsigned)(gh * WSz) * Ww +
                 (unsigned)(gw * WSz)) * 384ull);
    tdm_load_tile_2B(lds_addr_of(&sQ[head][0]), base + 2ull * (unsigned)ho,
                     32u, 7u, 7u, 384ull, (unsigned long long)Ww * 384ull);
    tdm_load_tile_2B(lds_addr_of(&sK[head][0]), base + 2ull * (unsigned)(128 + ho),
                     32u, 7u, 7u, 384ull, (unsigned long long)Ww * 384ull);
  }
  __builtin_amdgcn_s_wait_tensorcnt(0);

  const float scale = 0.17677669529663687f;  // 1/sqrt(32)
  const int jn = lane & 15, half = lane >> 4;

  for (int mq = 0; mq < 4; ++mq) {
    __syncthreads();  // pins LDS store/TDM -> load ordering (uniform control flow)
    // S row-block (16x64) = Q[mq] (16x32) @ K^T (32x64)
    v8f s[4] = {};
    v16bf aq = load_a_frag(sQ[head], 32, mq * 16, 0, lane);
#pragma unroll
    for (int t = 0; t < 4; ++t) {
      v16bf bk = load_b_frag(sK[head], 32, t * 16, 0, lane);
      s[t] = wmma_bf16(aq, bk, s[t]);
    }
    // scale + mask padded keys (j >= 49)
#pragma unroll
    for (int t = 0; t < 4; ++t) {
      bool bad = (16 * t + jn) >= WIN;
#pragma unroll
      for (int r = 0; r < 8; ++r) {
        float v = s[t][r] * scale;
        s[t][r] = bad ? -1e30f : v;
      }
    }
    // softmax over 64 key slots: in-register across 4 tiles, shuffle across 16 lanes
#pragma unroll
    for (int r = 0; r < 8; ++r) {
      float v = fmaxf(fmaxf(s[0][r], s[1][r]), fmaxf(s[2][r], s[3][r]));
      v = fmaxf(v, __shfl_xor(v, 1, 32));
      v = fmaxf(v, __shfl_xor(v, 2, 32));
      v = fmaxf(v, __shfl_xor(v, 4, 32));
      v = fmaxf(v, __shfl_xor(v, 8, 32));
      float sum = 0.f;
#pragma unroll
      for (int t = 0; t < 4; ++t) {
        float e = __expf(s[t][r] - v);
        s[t][r] = e;
        sum += e;
      }
      sum += __shfl_xor(sum, 1, 32);
      sum += __shfl_xor(sum, 2, 32);
      sum += __shfl_xor(sum, 4, 32);
      sum += __shfl_xor(sum, 8, 32);
      float inv = 1.0f / sum;
#pragma unroll
      for (int t = 0; t < 4; ++t)
        sP[head][(r + 8 * half) * 64 + 16 * t + jn] = f2bf(s[t][r] * inv);
    }
    __syncthreads();  // sP stores visible before A-fragment reads
    // O row-block (16x32) = P (16x64) @ V (64x32)
    v8f o[2] = {};
#pragma unroll
    for (int kt = 0; kt < 2; ++kt) {
      v16bf ap = load_a_frag(sP[head], 64, 0, kt * 32, lane);
#pragma unroll
      for (int nt = 0; nt < 2; ++nt) {
        v16bf bv = load_b_frag(sVT[head], 64, nt * 16, kt * 32, lane);
        o[nt] = wmma_bf16(ap, bv, o[nt]);
      }
    }
#pragma unroll
    for (int nt = 0; nt < 2; ++nt)
#pragma unroll
      for (int r = 0; r < 8; ++r) {
        int qrow = mq * 16 + r + 8 * half;
        if (qrow < WIN) {
          int h = gh * WSz + qrow / WSz;
          int w = gw * WSz + qrow % WSz;
          attnb[((size_t)b * Nn + h * Ww + w) * 128 + ho + nt * 16 + jn] =
              f2bf(o[nt][r]);
        }
      }
  }
}

// ---------------------------------------------------------------------------
// Kernel 4: out = (attn + v_lim) @ w_proj^T + b_proj   (M=200704, K=N=128)
// ---------------------------------------------------------------------------
__global__ __launch_bounds__(128) void proj_gemm(const unsigned short* __restrict__ a1,
                                                 const unsigned short* __restrict__ a2,
                                                 const float* __restrict__ w,
                                                 const float* __restrict__ bias,
                                                 float* __restrict__ out) {
  __shared__ unsigned short smA[64 * 128];
  __shared__ unsigned short smW[64 * 128];
  const int tid = threadIdx.x;
  const int n0 = blockIdx.x * 64;
  const int m0 = blockIdx.y * 64;
  for (int i = tid; i < 64 * 128; i += 128) {
    int r = i >> 7, k = i & 127;
    size_t ai = (size_t)(m0 + r) * 128 + k;
    smA[i] = f2bf(bf2f(a1[ai]) + bf2f(a2[ai]));
    smW[i] = f2bf(w[(size_t)(n0 + r) * 128 + k]);
  }
  __syncthreads();
  const int wave = tid >> 5, lane = tid & 31;
  v8f acc[4] = {};
#pragma unroll
  for (int kt = 0; kt < 4; ++kt) {
    v16bf bf = load_b_frag(smW, 128, wave * 16, kt * 32, lane);
#pragma unroll
    for (int mi = 0; mi < 4; ++mi) {
      v16bf af = load_a_frag(smA, 128, mi * 16, kt * 32, lane);
      acc[mi] = wmma_bf16(af, bf, acc[mi]);
    }
  }
  const int jn = lane & 15, half = lane >> 4;
  const int col = n0 + wave * 16 + jn;
  const float bcol = bias[col];
#pragma unroll
  for (int mi = 0; mi < 4; ++mi)
#pragma unroll
    for (int r = 0; r < 8; ++r) {
      int row = m0 + mi * 16 + r + 8 * half;
      out[(size_t)row * 128 + col] = acc[mi][r] + bcol;
    }
}

// ---------------------------------------------------------------------------
extern "C" void kernel_launch(void* const* d_in, const int* in_sizes, int n_in,
                              void* d_out, int out_size, void* d_ws, size_t ws_size,
                              hipStream_t stream) {
  const float* x      = (const float*)d_in[0];   // (B, N, C)
  const float* w_qkv  = (const float*)d_in[1];   // (3C, C)
  const float* w_lim  = (const float*)d_in[2];   // (C, 1, 3, 3)
  const float* b_lim  = (const float*)d_in[3];   // (C,)
  const float* w_proj = (const float*)d_in[4];   // (C, C)
  const float* b_proj = (const float*)d_in[5];   // (C,)
  float* out = (float*)d_out;                    // (B, N, C) fp32

  // workspace layout (bf16 stored as ushort):
  //   qkv   : Mtot * 384   (154.1 MB)
  //   attnb : Mtot * 128   ( 51.4 MB)
  //   vlim  : Mtot * 128   ( 51.4 MB)
  unsigned short* qkv   = (unsigned short*)d_ws;
  unsigned short* attnb = qkv + (size_t)Mtot * 384;
  unsigned short* vlim  = attnb + (size_t)Mtot * 128;

  qkv_gemm<<<dim3(6, Mtot / 64), 128, 0, stream>>>(x, w_qkv, qkv);
  lim_conv<<<(Mtot * 128) / 256, 256, 0, stream>>>(qkv, w_lim, b_lim, vlim);
  win_attn<<<dim3(64, Bb), 128, 0, stream>>>(qkv, attnb);
  proj_gemm<<<dim3(2, Mtot / 64), 128, 0, stream>>>(attnb, vlim, w_proj, b_proj, out);
}
